// MGU_79766132621952
// MI455X (gfx1250) — compile-verified, matrix operand order checked by reading, not run on hardware
//
#include <hip/hip_runtime.h>
#include <hip/hip_bf16.h>

// ---------------- problem constants ----------------
#define Bsz 32
#define Tsz 1024
#define Fsz 256
#define Hsz 512
#define Gsz 1024          // 2*H
#define NWG 32            // persistent workgroups in scan kernel (H/16)

typedef __attribute__((ext_vector_type(16))) __bf16 v16bf;
typedef __attribute__((ext_vector_type(8)))  __bf16 v8bf;
typedef __attribute__((ext_vector_type(8)))  float  v8f;
typedef __attribute__((ext_vector_type(4)))  unsigned int u32x4;
typedef __attribute__((ext_vector_type(8)))  int i32x8;
typedef __attribute__((ext_vector_type(4)))  int i32x4;

// Load a 16-half bf16 fragment: K = base+0..7 and base+16..23 of this lane's
// row, per the ISA 16-bit 16x32 A/B layout (two 16B vector loads).
__device__ __forceinline__ v16bf load_frag_bf16(const __bf16* __restrict__ p) {
    const v8bf a0 = *(const v8bf*)(p);
    const v8bf a1 = *(const v8bf*)(p + 16);
    v16bf v;
#pragma unroll
    for (int i = 0; i < 8; ++i) { v[i] = a0[i]; v[8 + i] = a1[i]; }
    return v;
}

// ---------------- TDM: 2D fp32 tile -> LDS (ISA 08_async_tensor §8) -------
// D# group0/group1 built per ISA bit layout; groups 2/3 zero (<=2D tensor).
// This toolchain exposes the 6-arg clang-23 builtin.
__device__ __forceinline__ void tdm_load_rows_f32(unsigned lds_byte_off,
                                                  const float* gptr,
                                                  unsigned nrows,
                                                  unsigned rowlen_elems) {
    unsigned long long ga = (unsigned long long)(const void*)gptr;
    u32x4 g0;
    g0[0] = 0x1u;                                         // count=1, user D#
    g0[1] = lds_byte_off;                                 // lds_addr (bytes)
    g0[2] = (unsigned)ga;                                 // global_addr[31:0]
    g0[3] = (unsigned)((ga >> 32) & 0x01FFFFFFu) | (2u << 30); // ga[56:32] | type=2
    i32x8 g1;
    g1[0] = (int)(2u << 16);                              // data_size=2 (4 bytes)
    g1[1] = (int)((rowlen_elems & 0xFFFFu) << 16);        // tensor_dim0[15:0]
    g1[2] = (int)(((rowlen_elems >> 16) & 0xFFFFu) |
                  ((nrows & 0xFFFFu) << 16));             // dim0 hi | tensor_dim1 lo
    g1[3] = (int)(((nrows >> 16) & 0xFFFFu) |
                  ((rowlen_elems & 0xFFFFu) << 16));      // dim1 hi | tile_dim0
    g1[4] = (int)(nrows & 0xFFFFu);                       // tile_dim1 (tile_dim2=0)
    g1[5] = (int)rowlen_elems;                            // tensor_dim0_stride[31:0]
    g1[6] = 0;
    g1[7] = 0;
    i32x4 z4 = {0, 0, 0, 0};
    i32x8 z8 = {0, 0, 0, 0, 0, 0, 0, 0};
    __builtin_amdgcn_tensor_load_to_lds(g0, g1, z4, z4, z8, 0);
}

// =====================================================================
// Kernel 0: fp32 -> bf16 pack (X and W_ih), memory-bound one-shot.
// =====================================================================
__global__ __launch_bounds__(256) void pack_bf16(const float* __restrict__ src,
                                                 __bf16* __restrict__ dst,
                                                 int nelem8) {   // nelem/8
    const int i = (blockIdx.x * 256 + threadIdx.x);
    if (i < nelem8) {
        const float4 a = *(const float4*)(src + (size_t)i * 8);
        const float4 b = *(const float4*)(src + (size_t)i * 8 + 4);
        v8bf v;
        v[0] = (__bf16)a.x; v[1] = (__bf16)a.y; v[2] = (__bf16)a.z; v[3] = (__bf16)a.w;
        v[4] = (__bf16)b.x; v[5] = (__bf16)b.y; v[6] = (__bf16)b.z; v[7] = (__bf16)b.w;
        *(v8bf*)(dst + (size_t)i * 8) = v;
    }
}

// =====================================================================
// Kernel 1: gi[t,b,g] = sum_f x[b,t,f] * W_ih[g,f] + b_ih[g]
// M = B*T = 32768, N = 1024, K = 256.  Pre-packed bf16 operands, fp32 acc.
// One wave computes a 16(M) x 64(N) strip (4 C-tiles, A-frag reused 4x);
// the 4 waves of a block share the same M-tile (A rows hit in cache).
// gi is stored time-major [T][B][G] for contiguous per-step reads in phase 2.
// =====================================================================
__global__ __launch_bounds__(128) void mgu_gi_gemm(
    const __bf16* __restrict__ xb, const __bf16* __restrict__ wb,
    const float* __restrict__ b_ih, float* __restrict__ gi) {
    const int lane  = threadIdx.x & 31;
    const int widx  = threadIdx.x >> 5;
    const int wid   = blockIdx.x * 4 + widx;
    const int mt    = wid >> 4;           // 0..2047  M-tile
    const int n0    = (wid & 15) * 64;    // N strip base
    const int lrow  = lane & 15;
    const int khalf = (lane >> 4) * 8;    // lanes 16-31 cover K+8 per ISA layout

    const __bf16* arow = xb + (size_t)(mt * 16 + lrow) * Fsz + khalf;

    v8f acc[4];
#pragma unroll
    for (int j = 0; j < 4; ++j) {
        const float bv = b_ih[n0 + j * 16 + lrow];   // C element n = lane&15
#pragma unroll
        for (int r = 0; r < 8; ++r) acc[j][r] = bv;
    }

#pragma unroll
    for (int kk = 0; kk < 8; ++kk) {
        const int kb = kk * 32;
        const v16bf a = load_frag_bf16(arow + kb);
#pragma unroll
        for (int j = 0; j < 4; ++j) {
            const v16bf b =
                load_frag_bf16(wb + (size_t)(n0 + j * 16 + lrow) * Fsz + khalf + kb);
            acc[j] = __builtin_amdgcn_wmma_f32_16x16x32_bf16(
                false, a, false, b, (short)0, acc[j], false, false);
        }
    }

#pragma unroll
    for (int j = 0; j < 4; ++j) {
        const int n = n0 + j * 16 + lrow;
#pragma unroll
        for (int r = 0; r < 8; ++r) {
            const int m  = r + ((lane >> 4) << 3);
            const int rt = mt * 16 + m;            // row in [B*T]
            const int b  = rt >> 10;               // rt / T
            const int t  = rt & (Tsz - 1);         // rt % T
            gi[((size_t)t * Bsz + b) * Gsz + n] = acc[j][r];
        }
    }
}

// =====================================================================
// Kernel 2: persistent cooperative scan. 32 WGs x 64 threads (2 waves).
// WG owns hidden columns [j0, j0+16). W_hh rows for both gates are
// TDM-loaded to LDS once, converted to bf16 B-fragments. h ping-pongs
// through global (bf16, L2-resident); grid sync = generation counter.
// =====================================================================
__global__ __launch_bounds__(64, 1) void mgu_scan(
    const float* __restrict__ gi, const float* __restrict__ w_hh,
    const float* __restrict__ b_hh, float* __restrict__ out,
    __bf16* __restrict__ hbuf, unsigned* __restrict__ sync) {
    __shared__ float  ldsW[16 * Hsz];          // 32 KB fp32 staging (one gate)
    __shared__ __bf16 ldsB[2][16][32][16];     // 32 KB B-fragments, both gates

    const int j0    = blockIdx.x * 16;
    const int lane  = threadIdx.x & 31;
    const int mt    = threadIdx.x >> 5;        // wave -> M-tile (rows 16*mt..)
    const int lrow  = lane & 15;
    const int khalf = (lane >> 4) * 8;

    // ---- one-time: stage W_hh slices via Tensor Data Mover ----
    const unsigned lds0 = (unsigned)(unsigned long long)(&ldsW[0]);
    for (int g = 0; g < 2; ++g) {
        if (threadIdx.x < 32) {   // wave 0 issues the DMA
            tdm_load_rows_f32(lds0, w_hh + (size_t)(g * Hsz + j0) * Hsz, 16, Hsz);
            __builtin_amdgcn_s_wait_tensorcnt(0);
        }
        __syncthreads();
        // convert staged fp32 rows -> bf16 WMMA B-fragment layout
        for (int kk = mt; kk < 16; kk += 2) {
#pragma unroll
            for (int p = 0; p < 16; ++p) {
                const int k = kk * 32 + khalf + (p < 8 ? p : p + 8);
                ldsB[g][kk][lane][p] = (__bf16)ldsW[lrow * Hsz + k];
            }
        }
        __syncthreads();
    }

    const int   n     = j0 + lrow;             // this lane's output column
    const float bhf   = b_hh[n];
    const float bhn   = b_hh[Hsz + n];
    const int   mrow0 = mt * 16 + ((lane >> 4) << 3);   // batch row of acc[0]

    for (int t = 0; t < Tsz; ++t) {
        const __bf16* hrow =
            hbuf + (size_t)(t & 1) * (Bsz * Hsz) + (size_t)(mt * 16 + lrow) * Hsz + khalf;
        const float* gslab = gi + (size_t)t * Bsz * Gsz;   // contiguous [B][G]

        v8f accf, accn;
#pragma unroll
        for (int r = 0; r < 8; ++r) {          // fold gi_f + bias_hh_f into C
            accf[r] = gslab[(size_t)(mrow0 + r) * Gsz + n] + bhf;
            accn[r] = bhn;
        }

        // prefetch next timestep's gi slab while we grind WMMAs / wait at sync
        if (t + 1 < Tsz) {
            const float* nf = gi + ((size_t)(t + 1) * Bsz + mrow0) * Gsz + n;
            __builtin_prefetch(nf, 0, 0);
            __builtin_prefetch(nf + Hsz, 0, 0);
        }

#pragma unroll 4
        for (int kk = 0; kk < 16; ++kk) {      // K = 512 in steps of 32
            const v16bf a  = load_frag_bf16(hrow + kk * 32);
            const v16bf bf = *(const v16bf*)&ldsB[0][kk][lane][0];
            const v16bf bn = *(const v16bf*)&ldsB[1][kk][lane][0];
            accf = __builtin_amdgcn_wmma_f32_16x16x32_bf16(
                false, a, false, bf, (short)0, accf, false, false);
            accn = __builtin_amdgcn_wmma_f32_16x16x32_bf16(
                false, a, false, bn, (short)0, accn, false, false);
        }

        __bf16* hnext = hbuf + (size_t)((t + 1) & 1) * (Bsz * Hsz);
#pragma unroll
        for (int r = 0; r < 8; ++r) {
            const int b     = mrow0 + r;
            const float gin = gslab[(size_t)b * Gsz + Hsz + n];
            const float f   = 1.0f / (1.0f + __expf(-accf[r]));
            const float hn  = tanhf(gin + f * accn[r]);
            out[(size_t)(b * Tsz + t) * Hsz + n] = hn;
            hnext[(size_t)b * Hsz + n] = (__bf16)hn;
        }

        // ---- grid-wide generation sync (32 WGs) ----
        __threadfence();
        __syncthreads();
        if (threadIdx.x == 0) {
            __hip_atomic_fetch_add(sync, 1u, __ATOMIC_RELEASE, __HIP_MEMORY_SCOPE_AGENT);
            const unsigned target = (unsigned)(t + 1) * NWG;
            while (__hip_atomic_load(sync, __ATOMIC_ACQUIRE, __HIP_MEMORY_SCOPE_AGENT) <
                   target)
                __builtin_amdgcn_s_sleep(1);
        }
        __syncthreads();
        __threadfence();
    }
}

// =====================================================================
extern "C" void kernel_launch(void* const* d_in, const int* in_sizes, int n_in,
                              void* d_out, int out_size, void* d_ws, size_t ws_size,
                              hipStream_t stream) {
    const float* input = (const float*)d_in[0];   // [32,1024,256]
    const float* w_ih  = (const float*)d_in[1];   // [1024,256]
    const float* w_hh  = (const float*)d_in[2];   // [1024,512]
    const float* b_ih  = (const float*)d_in[3];   // [1024]
    const float* b_hh  = (const float*)d_in[4];   // [1024]
    (void)d_in[5]; (void)in_sizes; (void)n_in; (void)out_size;

    float* out = (float*)d_out;                   // [32,1024,512] fp32

    // workspace: gi fp32 [T][B][G] (128 MB) | Xbf16 (16 MB) | Wbf16 (512 KB)
    //          | h double-buffer bf16 (64 KB) | sync counter
    const size_t GI_BYTES = (size_t)Tsz * Bsz * Gsz * sizeof(float);
    const size_t XB_BYTES = (size_t)Bsz * Tsz * Fsz * sizeof(__bf16);
    const size_t WB_BYTES = (size_t)Gsz * Fsz * sizeof(__bf16);
    const size_t HB_BYTES = (size_t)2 * Bsz * Hsz * sizeof(__bf16);
    char* ws = (char*)d_ws;
    float*    gi      = (float*)ws;
    __bf16*   xb      = (__bf16*)(ws + GI_BYTES);
    __bf16*   wb      = (__bf16*)(ws + GI_BYTES + XB_BYTES);
    __bf16*   hbuf    = (__bf16*)(ws + GI_BYTES + XB_BYTES + WB_BYTES);
    unsigned* counter = (unsigned*)(ws + GI_BYTES + XB_BYTES + WB_BYTES + HB_BYTES);
    (void)ws_size;

    // h0 = 0 and sync counter = 0 (capture-safe, re-done every replay)
    (void)hipMemsetAsync(hbuf, 0, HB_BYTES + sizeof(unsigned), stream);

    // Phase 0: one-shot bf16 packs of X and W_ih
    const int xn8 = (Bsz * Tsz * Fsz) / 8;        // 1,048,576
    const int wn8 = (Gsz * Fsz) / 8;              // 32,768
    pack_bf16<<<(xn8 + 255) / 256, 256, 0, stream>>>(input, xb, xn8);
    pack_bf16<<<(wn8 + 255) / 256, 256, 0, stream>>>(w_ih, wb, wn8);

    // Phase 1: gi GEMM — 2048 M-tiles x 16 N-strips, 4 waves/block
    mgu_gi_gemm<<<8192, 128, 0, stream>>>(xb, wb, b_ih, gi);

    // Phase 2: persistent scan — 32 WGs, 2 waves each
    mgu_scan<<<NWG, 64, 0, stream>>>(gi, w_hh, b_hh, out, hbuf, counter);
}
